// DINOPPNet_82892868812969
// MI455X (gfx1250) — compile-verified, matrix operand order checked by reading, not run on hardware
//
#include <hip/hip_runtime.h>
#include <cstdint>
#include <cstddef>

// ---------------------------------------------------------------------------
// Problem constants (DINOPPNet head)
//   patches    [B=64, N=256, D=1024] f32
//   prototypes [P=2000, D=1024]      f32
//   fc_w       [C=200, P=2000]       f32
// Outputs (concatenated in d_out, f32):
//   logits [B,C] (12800) | attn_maps [B,P,N] (32768000) | min_dist [B,P] (128000)
// Workspace: p2 (2000 f32, padded to 8192 B) | protoHi bf16[2000*1024] |
//            protoLo bf16[2000*1024]  => ~8.2 MB total.
// ---------------------------------------------------------------------------
constexpr int kB = 64;
constexpr int kN = 256;
constexpr int kD = 1024;
constexpr int kP = 2000;
constexpr int kC = 200;
constexpr float kEPS = 1e-4f;

typedef __attribute__((ext_vector_type(16))) __bf16 v16bf;
typedef __attribute__((ext_vector_type(8)))  float  v8f;
typedef __attribute__((ext_vector_type(4)))  unsigned int u32x4;
typedef __attribute__((ext_vector_type(8)))  int         i32x8;
typedef __attribute__((ext_vector_type(4)))  int         i32x4;

union ABReg {
    v16bf bf;
    unsigned int u[8];
};

// Round-to-nearest-even f32 -> bf16 (bits in low 16)
__device__ __forceinline__ unsigned int bf16_rne(float x) {
    unsigned int u = __float_as_uint(x);
    return (u + 0x7FFFu + ((u >> 16) & 1u)) >> 16;
}

// Split two floats into packed bf16 hi parts and packed bf16 lo (residual) parts.
__device__ __forceinline__ void split2(float a, float b, unsigned int& hp, unsigned int& lp) {
    unsigned int ha = bf16_rne(a);
    unsigned int hb = bf16_rne(b);
    float ra = a - __uint_as_float(ha << 16);
    float rb = b - __uint_as_float(hb << 16);
    unsigned int la = bf16_rne(ra);
    unsigned int lb = bf16_rne(rb);
    hp = ha | (hb << 16);
    lp = la | (lb << 16);
}

// ---------------------------------------------------------------------------
// Kernel 1: per prototype row -- p2[p] = sum_d w[p][d]^2, and pre-split the
// row into packed bf16 hi / lo arrays for TDM consumption by the main kernel.
// One block per prototype row; thread t owns elements 4t..4t+3.
// ---------------------------------------------------------------------------
__global__ __launch_bounds__(256, 1) void proto_prep(const float* __restrict__ prot,
                                                     float* __restrict__ p2,
                                                     unsigned int* __restrict__ protoHiU,
                                                     unsigned int* __restrict__ protoLoU) {
    __shared__ float red[256];
    const int p = blockIdx.x;
    const int t = threadIdx.x;
    float4 v = *(const float4*)(prot + (size_t)p * kD + t * 4);  // 1024 = 256*4
    red[t] = v.x * v.x + v.y * v.y + v.z * v.z + v.w * v.w;

    uint2 h, l;
    split2(v.x, v.y, h.x, l.x);
    split2(v.z, v.w, h.y, l.y);
    ((uint2*)protoHiU)[(size_t)p * 256 + t] = h;   // row-major bf16 pairs along D
    ((uint2*)protoLoU)[(size_t)p * 256 + t] = l;

    __syncthreads();
    for (int off = 128; off > 0; off >>= 1) {
        if (t < off) red[t] += red[t + off];
        __syncthreads();
    }
    if (t == 0) p2[p] = red[0];
}

// ---------------------------------------------------------------------------
// TDM helper: issue a 2-D tensor_load_to_lds of an 80-row x 32-element bf16
// tile (64 B per row) from a [2000 x 1024] bf16 tensor into LDS.
// D# per CDNA5 ISA 8.3-8.5. Groups 2/3 (and the trailing group) zero: 2-D tile.
// This toolchain's builtin takes 6 args:
//   (u32x4 g0, i32x8 g1, i32x4, i32x4, i32x8, i32 cpol)
// ---------------------------------------------------------------------------
__device__ __forceinline__ void tdm_load_tile_bf16(const unsigned int* srcU,
                                                   unsigned int ldsByteAddr,
                                                   int p0, int k0) {
    const unsigned long long gaddr =
        (unsigned long long)(uintptr_t)srcU + (((unsigned long long)p0 * kD + k0) * 2ull);

    u32x4 g0;
    g0[0] = 1u;                                   // count=1, is_restore=0, gather off
    g0[1] = ldsByteAddr;                          // lds_addr
    g0[2] = (unsigned int)(gaddr & 0xffffffffull);
    g0[3] = (unsigned int)((gaddr >> 32) & 0x01ffffffull) | (2u << 30);  // type=2

    i32x8 g1;
    g1[0] = (int)(1u << 16);                      // workgroup_mask=0, data_size=1 (2 B)
    g1[1] = (int)(((unsigned)kD & 0xffffu) << 16);      // tensor_dim0 = 1024 (lo16)
    g1[2] = (int)(((unsigned)kP & 0xffffu) << 16);      // dim0 hi=0 | tensor_dim1 = 2000
    g1[3] = (int)(32u << 16);                     // tensor_dim1 hi=0 | tile_dim0 = 32
    g1[4] = 80;                                   // tile_dim1 = 80, tile_dim2 = 0
    g1[5] = kD;                                   // tensor_dim0_stride = 1024 elems
    g1[6] = 0;                                    // stride hi, tensor_dim1_stride
    g1[7] = 0;

    i32x4 gz4 = {0, 0, 0, 0};
    i32x8 gz8 = {0, 0, 0, 0, 0, 0, 0, 0};
    __builtin_amdgcn_tensor_load_to_lds(g0, g1, gz4, gz4, gz8, 0);
}

// ---------------------------------------------------------------------------
// Kernel 2: main fused GEMM + distance epilogue.
//   grid.x = 25 (column groups of 80 prototypes = 5 WMMA tiles)
//   grid.y = 64 (batch)
//   block  = 256 threads = 8 waves; wave w owns patch rows [32w, 32w+32)
// Split-bf16 GEMM: xp ~= Ahi*Bhi + Ahi*Blo + Alo*Bhi, f32 accum via
// v_wmma_f32_16x16x32_bf16. B tiles arrive in LDS via the Tensor Data Mover.
// ---------------------------------------------------------------------------
__global__ __launch_bounds__(256, 1) void ppnet_main(const float* __restrict__ patches,
                                                     const unsigned int* __restrict__ protoHiU,
                                                     const unsigned int* __restrict__ protoLoU,
                                                     const float* __restrict__ p2g,
                                                     float* __restrict__ attn,
                                                     float* __restrict__ mind) {
    // Packed bf16 pairs (uint = 2 bf16 along K). Chunk = 32 K values = 16 uints/row.
    __shared__ unsigned int aHiU[kN * 16];   // 16 KB
    __shared__ unsigned int aLoU[kN * 16];   // 16 KB
    __shared__ unsigned int bHiU[80 * 16];   //  5 KB  (TDM destination)
    __shared__ unsigned int bLoU[80 * 16];   //  5 KB  (TDM destination)
    __shared__ float x2s[kN];
    __shared__ int   minb[80];

    const int t    = threadIdx.x;
    const int lane = t & 31;
    const int wave = t >> 5;
    const int b    = blockIdx.y;
    const int p0   = blockIdx.x * 80;

    // Generic->LDS: low 32 bits of the generic address are the LDS byte offset.
    const unsigned int bHiLds = (unsigned int)(uintptr_t)&bHiU[0];
    const unsigned int bLoLds = (unsigned int)(uintptr_t)&bLoU[0];

    const v8f zero8 = {0.f, 0.f, 0.f, 0.f, 0.f, 0.f, 0.f, 0.f};
    v8f acc[2][5];
#pragma unroll
    for (int m = 0; m < 2; ++m)
#pragma unroll
        for (int j = 0; j < 5; ++j) acc[m][j] = zero8;

    float x2acc = 0.f;
    const float* arow = patches + ((size_t)b * kN + t) * kD;  // row t of this batch

    // WMMA 16-bit A/B per-lane K layout: lanes 0-15 hold K {0..7,16..23},
    // lanes 16-31 hold K {8..15,24..31} (uint offset 0 / 4, second seg +8).
    const int segsel = (lane < 16) ? 0 : 4;
    const int rowA   = wave * 32 + (lane & 15);
    const int colB   = lane & 15;

    for (int k0 = 0; k0 < kD; k0 += 32) {
        // --- Kick off B-tile DMA (80 x 32 bf16, hi & lo) via the TDM ---
        if (wave == 0) {
            tdm_load_tile_bf16(protoHiU, bHiLds, p0, k0);
            tdm_load_tile_bf16(protoLoU, bLoLds, p0, k0);
        }

        // --- Stage A: thread t converts its own row's 32 elements ---
#pragma unroll
        for (int q = 0; q < 8; ++q) {
            float4 v = *(const float4*)(arow + k0 + q * 4);
            x2acc += v.x * v.x + v.y * v.y + v.z * v.z + v.w * v.w;
            unsigned int h0, l0, h1, l1;
            split2(v.x, v.y, h0, l0);
            split2(v.z, v.w, h1, l1);
            aHiU[t * 16 + q * 2]     = h0;
            aHiU[t * 16 + q * 2 + 1] = h1;
            aLoU[t * 16 + q * 2]     = l0;
            aLoU[t * 16 + q * 2 + 1] = l1;
        }
        if (k0 + 32 < kD) __builtin_prefetch(arow + k0 + 32, 0, 0);  // global_prefetch_b8

        // Wave 0 waits for its DMAs (no-op for other waves), then all sync.
        __builtin_amdgcn_s_wait_tensorcnt(0);
        __syncthreads();

        // --- Compute: load fragments from LDS, 30 WMMA per wave per chunk ---
        ABReg aH[2], aL[2];
#pragma unroll
        for (int m = 0; m < 2; ++m) {
            const int base = (rowA + m * 16) * 16 + segsel;
            *(uint4*)&aH[m].u[0] = *(const uint4*)&aHiU[base];
            *(uint4*)&aH[m].u[4] = *(const uint4*)&aHiU[base + 8];
            *(uint4*)&aL[m].u[0] = *(const uint4*)&aLoU[base];
            *(uint4*)&aL[m].u[4] = *(const uint4*)&aLoU[base + 8];
        }
#pragma unroll
        for (int j = 0; j < 5; ++j) {
            const int bbase = (j * 16 + colB) * 16 + segsel;
            ABReg bH, bL;
            *(uint4*)&bH.u[0] = *(const uint4*)&bHiU[bbase];
            *(uint4*)&bH.u[4] = *(const uint4*)&bHiU[bbase + 8];
            *(uint4*)&bL.u[0] = *(const uint4*)&bLoU[bbase];
            *(uint4*)&bL.u[4] = *(const uint4*)&bLoU[bbase + 8];
#pragma unroll
            for (int m = 0; m < 2; ++m) {
                acc[m][j] = __builtin_amdgcn_wmma_f32_16x16x32_bf16(
                    false, aH[m].bf, false, bH.bf, (short)0, acc[m][j], false, false);
                acc[m][j] = __builtin_amdgcn_wmma_f32_16x16x32_bf16(
                    false, aH[m].bf, false, bL.bf, (short)0, acc[m][j], false, false);
                acc[m][j] = __builtin_amdgcn_wmma_f32_16x16x32_bf16(
                    false, aL[m].bf, false, bH.bf, (short)0, acc[m][j], false, false);
            }
        }
        __syncthreads();
    }

    // --- Epilogue: dist = relu(x2 - 2*xp + p2), transposed store, min over n ---
    x2s[t] = x2acc;
    if (t < 80) minb[t] = 0x7F800000;  // +inf
    __syncthreads();

    const int hi = lane >> 4;  // C layout: VGPR r holds row M = r + 8*hi, col N = lane&15
#pragma unroll
    for (int j = 0; j < 5; ++j) {
        const int pcol = p0 + j * 16 + colB;
        const float p2v = p2g[pcol];
#pragma unroll
        for (int m = 0; m < 2; ++m) {
            const int nbase = wave * 32 + m * 16 + hi * 8;
            float d[8];
            float mn = 3.4e38f;
#pragma unroll
            for (int r = 0; r < 8; ++r) {
                float dist = x2s[nbase + r] - 2.f * acc[m][j][r] + p2v;
                dist = fmaxf(dist, 0.f);
                d[r] = dist;
                mn = fminf(mn, dist);
            }
            float* dst = attn + ((size_t)b * kP + pcol) * kN + nbase;
            *(float4*)(dst)     = make_float4(d[0], d[1], d[2], d[3]);
            *(float4*)(dst + 4) = make_float4(d[4], d[5], d[6], d[7]);
            // dist >= 0 ==> float ordering == int ordering on the raw bits
            atomicMin(&minb[j * 16 + colB], __float_as_int(mn));
        }
    }
    __syncthreads();
    if (t < 80) mind[(size_t)b * kP + p0 + t] = __int_as_float(minb[t]);
}

// ---------------------------------------------------------------------------
// Kernel 3: activations + FC.  logits[b,c] = sum_p log((m+1)/(m+eps)) * fc_w[c,p]
// ---------------------------------------------------------------------------
__global__ __launch_bounds__(256, 1) void logits_kernel(const float* __restrict__ mind,
                                                        const float* __restrict__ fc_w,
                                                        float* __restrict__ logits) {
    __shared__ float act[kP];
    const int b = blockIdx.x;
    const int t = threadIdx.x;
    for (int p = t; p < kP; p += 256) {
        float m = mind[(size_t)b * kP + p];
        act[p] = logf((m + 1.0f) / (m + kEPS));
    }
    __syncthreads();
    for (int c = t; c < kC; c += 256) {
        const float* w = fc_w + (size_t)c * kP;
        float s = 0.f;
        for (int p = 0; p < kP; p += 4) {
            float4 wv = *(const float4*)(w + p);
            s += act[p] * wv.x + act[p + 1] * wv.y + act[p + 2] * wv.z + act[p + 3] * wv.w;
        }
        logits[(size_t)b * kC + c] = s;
    }
}

// ---------------------------------------------------------------------------
extern "C" void kernel_launch(void* const* d_in, const int* in_sizes, int n_in,
                              void* d_out, int out_size, void* d_ws, size_t ws_size,
                              hipStream_t stream) {
    (void)in_sizes; (void)n_in; (void)out_size; (void)ws_size;
    const float* patches    = (const float*)d_in[0];  // [64,256,1024]
    const float* prototypes = (const float*)d_in[1];  // [2000,1024]
    const float* fc_w       = (const float*)d_in[2];  // [200,2000]

    float* out        = (float*)d_out;
    float* out_logits = out;                                   // 12800
    float* out_attn   = out + (size_t)kB * kC;                 // 32768000
    float* out_min    = out_attn + (size_t)kB * kP * kN;       // 128000

    // Workspace: [p2: 8192 B][protoHi: 4 MB][protoLo: 4 MB]
    char* ws = (char*)d_ws;
    float* p2              = (float*)ws;
    unsigned int* protoHiU = (unsigned int*)(ws + 8192);
    unsigned int* protoLoU = (unsigned int*)(ws + 8192 + (size_t)kP * kD * 2);

    proto_prep<<<dim3(kP), dim3(256), 0, stream>>>(prototypes, p2, protoHiU, protoLoU);
    ppnet_main<<<dim3(kP / 80, kB), dim3(256), 0, stream>>>(patches, protoHiU, protoLoU,
                                                            p2, out_attn, out_min);
    logits_kernel<<<dim3(kB), dim3(256), 0, stream>>>(out_min, fc_w, out_logits);
}
